// SequentialMNIST_Vanilla_SNN_2637109920477
// MI455X (gfx1250) — compile-verified, matrix operand order checked by reading, not run on hardware
//
#include <hip/hip_runtime.h>

typedef __attribute__((ext_vector_type(16))) _Float16 v16h;
typedef __attribute__((ext_vector_type(8)))  float    v8f;

#define T_STEPS  784
#define T_START  522          // 784*2//3
#define H_DIM    128
#define OUT_DIM  10
#define TILE_STR 17           // 16 cols + 1 pad to break LDS bank conflicts
#define KB       8            // timesteps per barrier block
#define NBLK     (T_STEPS / KB)   // 98

// K-slot mapping for the f16 16x16x32 WMMA A operand (mirrored for B):
// lanes 0-15 hold K in {0..7, 16..23}; lanes 16-31 hold K in {8..15, 24..31};
// vector element j (2 halves per VGPR, ascending) -> local K index.
__device__ __forceinline__ int kmap(int j, int hsel) {
    return (j < 8 ? j : j + 8) + (hsel << 3);
}

__global__ __launch_bounds__(128)
void snn_scan_wmma_kernel(const float* __restrict__ x,
                          const float* __restrict__ W1,
                          const float* __restrict__ b1,
                          const float* __restrict__ W2,
                          const float* __restrict__ b2,
                          float* __restrict__ out)
{
    // double-buffered block of per-step h2 tiles (16x16 f32, padded stride)
    __shared__ float tile[2][KB][16 * TILE_STR];

    const int tid  = threadIdx.x;
    const int wid  = tid >> 5;          // wave id 0..3 -> 32-channel H chunk
    const int lane = tid & 31;
    const int col  = lane & 15;         // M row (A) / N col (B, C/D)
    const int hsel = lane >> 4;
    const int rowBase = blockIdx.x * 16;

    // ---- per-lane resident constants & state ----
    float v1[16], hW1[16], hb1[16];
    v16h bmat;
    #pragma unroll
    for (int j = 0; j < 16; ++j) {
        int h = (wid << 5) + kmap(j, hsel);
        hW1[j] = 0.5f * W1[h];          // fold 1/tau = 0.5 into the weights
        hb1[j] = 0.5f * b1[h];
        v1[j]  = 0.0f;
        bmat[j] = (col < OUT_DIM) ? (_Float16)W2[col * H_DIM + h] : (_Float16)0.0f;
    }
    const float b2v = (col < OUT_DIM) ? b2[col] : 0.0f;

    float v2[8], s2acc[8];
    #pragma unroll
    for (int r = 0; r < 8; ++r) { v2[r] = 0.0f; s2acc[r] = 0.0f; }

    // C/D layout: vector element r <-> (M = r + 8*hsel, N = col)
    const int cdoff = hsel * 8 * TILE_STR + col;

    const float*  xrow = x + (size_t)(rowBase + col) * T_STEPS;
    const float4* xv4  = reinterpret_cast<const float4*>(xrow);

    // seed all tiles with the bias (h2 = b2 + sum of wave partials);
    // wave0 seeds phase 0, wave1 seeds phase 1.
    if (wid < 2) {
        #pragma unroll
        for (int s = 0; s < KB; ++s)
            #pragma unroll
            for (int r = 0; r < 8; ++r)
                tile[wid][s][cdoff + r * TILE_STR] = b2v;
    }
    __syncthreads();

    #pragma unroll 1
    for (int blk = 0; blk < NBLK; ++blk) {
        const int phase = blk & 1;

        float4 xa = xv4[2 * blk];
        float4 xb = xv4[2 * blk + 1];
        float xs[KB] = { xa.x, xa.y, xa.z, xa.w, xb.x, xb.y, xb.z, xb.w };
        __builtin_prefetch(xrow + blk * KB + 64, 0, 0);   // stream x ahead

        // 8 timesteps of LIF1 + WMMA, no sync: only the v1 FMA chain is serial,
        // so 8 independent WMMAs overlap.
        #pragma unroll
        for (int s = 0; s < KB; ++s) {
            float xv = xs[s];
            v16h a;
            #pragma unroll
            for (int j = 0; j < 16; ++j) {
                float vn = fmaf(xv, hW1[j], fmaf(0.5f, v1[j], hb1[j]));
                bool sp = vn >= 1.0f;
                v1[j] = sp ? 0.0f : vn;                // hard reset
                a[j]  = sp ? (_Float16)1.0f : (_Float16)0.0f;
            }

            // partial h2 for this K chunk: [16x32] x [32x16] -> 16x16 f32
            v8f c = {};
            c = __builtin_amdgcn_wmma_f32_16x16x32_f16(false, a, false, bmat,
                                                       (short)0, c, false, false);

            float* cur = &tile[phase][s][0];
            #pragma unroll
            for (int r = 0; r < 8; ++r)
                atomicAdd(&cur[cdoff + r * TILE_STR], c[r]);   // ds_add_f32
        }

        __syncthreads();        // one barrier per 8 timesteps

        if (wid == 0) {
            // replay the 8 buffered steps of LIF2 sequentially, re-seeding
            // each consumed tile with b2 for reuse two blocks from now.
            #pragma unroll
            for (int s = 0; s < KB; ++s) {
                const int t = blk * KB + s;
                float* cur = &tile[phase][s][0];
                #pragma unroll
                for (int r = 0; r < 8; ++r) {
                    float h2 = cur[cdoff + r * TILE_STR];
                    float vn = 0.5f * v2[r] + 0.5f * h2;
                    bool sp = vn >= 1.0f;
                    v2[r] = sp ? 0.0f : vn;
                    if (t >= T_START) s2acc[r] += sp ? 1.0f : 0.0f;
                    cur[cdoff + r * TILE_STR] = b2v;   // re-seed
                }
            }
        }
    }

    if (wid == 0 && col < OUT_DIM) {
        #pragma unroll
        for (int r = 0; r < 8; ++r) {
            int m = r + 8 * hsel;
            out[(size_t)(rowBase + m) * OUT_DIM + col] = s2acc[r];
        }
    }
}

extern "C" void kernel_launch(void* const* d_in, const int* in_sizes, int n_in,
                              void* d_out, int out_size, void* d_ws, size_t ws_size,
                              hipStream_t stream) {
    const float* x  = (const float*)d_in[0];   // [4096, 784, 1]
    const float* W1 = (const float*)d_in[1];   // [128, 1]
    const float* b1 = (const float*)d_in[2];   // [128]
    const float* W2 = (const float*)d_in[3];   // [10, 128]
    const float* b2 = (const float*)d_in[4];   // [10]
    float* out = (float*)d_out;                // [4096, 10]

    dim3 grid(4096 / 16);
    dim3 block(128);
    hipLaunchKernelGGL(snn_scan_wmma_kernel, grid, block, 0, stream,
                       x, W1, b1, W2, b2, out);
}
